// OELMMultiHeadAttention_34050500722760
// MI455X (gfx1250) — compile-verified, matrix operand order checked by reading, not run on hardware
//
#include <hip/hip_runtime.h>

// ---------------------------------------------------------------------------
// MI455X (gfx1250) fused causal MHA forward, bf16 WMMA pipeline.
//   B=2, S=2048, D=1024, H=16, Hd=64
// v2: flash-attention q-tile 32 rows/wave (halves K/V L2 traffic, 2x wmma
//     per vmem load in the attention loop); near-scope prefetches.
// ---------------------------------------------------------------------------

#define B_ 2
#define S_ 2048
#define D_ 1024
#define H_ 16
#define HD_ 64
#define M_ (B_ * S_)   // 4096 rows

typedef __attribute__((ext_vector_type(16))) __bf16 v16bf;
typedef __attribute__((ext_vector_type(8)))  float  v8f;

#define WMMA_BF16(a, b, c) \
  __builtin_amdgcn_wmma_f32_16x16x32_bf16(false, (a), false, (b), (short)0, (c), false, false)

#define NEG_INF (-3.402823466e38f)

__device__ __forceinline__ unsigned short f32_to_bf16(float f) {
  unsigned int u = __float_as_uint(f);
  u += 0x7fffu + ((u >> 16) & 1u);   // round-to-nearest-even
  return (unsigned short)(u >> 16);
}

__device__ __forceinline__ v8f vzero8() {
  v8f v = {0.f, 0.f, 0.f, 0.f, 0.f, 0.f, 0.f, 0.f};
  return v;
}

// NT fragment loader: for A (16x32, row m = lane&15) and, by N<->M duality,
// for B from a row-major [N][K] matrix (both operands K-contiguous).
// Lane l: m = l&15, ksel = l>>4; elems 0..7 = [m][k0+ksel*8 .. +7],
// elems 8..15 = [m][k0+ksel*8+16 .. +23].   (ISA 7.12.2, 16-bit A 16x32)
__device__ __forceinline__ v16bf load_frag_nt(const unsigned short* __restrict__ p,
                                              int ld, int row0, int k0, int lane) {
  int m = lane & 15;
  int sel = lane >> 4;
  const unsigned short* q = p + ((size_t)(row0 + m)) * (size_t)ld + k0 + sel * 8;
  union { v16bf v; uint4 u[2]; } f;
  f.u[0] = *(const uint4*)(q);
  f.u[1] = *(const uint4*)(q + 16);
  return f.v;
}

// ---------------------------------------------------------------------------
// f32 -> bf16 conversion
// ---------------------------------------------------------------------------
__global__ void k_f32_to_bf16(const float* __restrict__ in,
                              unsigned short* __restrict__ out, int n) {
  int i = blockIdx.x * blockDim.x + threadIdx.x;
  if (i < n) out[i] = f32_to_bf16(in[i]);
}

// ---------------------------------------------------------------------------
// Shared GEMM core: C[128x128 block] = A[M x K] * W[N x K]^T  (NT, bf16->f32)
// 8 waves: wave tile 64(M) x 32(N) = 4x2 fragments.
// ---------------------------------------------------------------------------
__device__ __forceinline__ void gemm_core(const unsigned short* __restrict__ A,
                                          const unsigned short* __restrict__ W,
                                          int Kdim, int m0, int n0, int lane,
                                          v8f acc[4][2]) {
#pragma unroll
  for (int i = 0; i < 4; ++i) {
    acc[i][0] = vzero8();
    acc[i][1] = vzero8();
  }
  for (int k0 = 0; k0 < Kdim; k0 += 32) {
    if (k0 + 32 < Kdim) {
      // speculative near-scope prefetch of next K-slab (global_prefetch_b8)
      __builtin_prefetch(A + ((size_t)(m0 + (lane & 15))) * (size_t)Kdim + k0 + 32, 0, 3);
      __builtin_prefetch(W + ((size_t)(n0 + (lane & 15))) * (size_t)Kdim + k0 + 32, 0, 3);
    }
    v16bf b0 = load_frag_nt(W, Kdim, n0,      k0, lane);
    v16bf b1 = load_frag_nt(W, Kdim, n0 + 16, k0, lane);
#pragma unroll
    for (int i = 0; i < 4; ++i) {
      v16bf a = load_frag_nt(A, Kdim, m0 + i * 16, k0, lane);
      acc[i][0] = WMMA_BF16(a, b0, acc[i][0]);
      acc[i][1] = WMMA_BF16(a, b1, acc[i][1]);
    }
  }
}

// ---------------------------------------------------------------------------
// QKV projection: q/k -> [B,H,S,Hd] bf16 ; v -> [B,H,Hd,S] bf16 (transposed)
// grid (M/128, D/128, 3), block 256.
// ---------------------------------------------------------------------------
__global__ __launch_bounds__(256) void
k_qkv_gemm(const unsigned short* __restrict__ xb,
           const unsigned short* __restrict__ wq,
           const unsigned short* __restrict__ wk,
           const unsigned short* __restrict__ wv,
           unsigned short* __restrict__ qb,
           unsigned short* __restrict__ kb,
           unsigned short* __restrict__ vt) {
  int lane = threadIdx.x & 31;
  int w = threadIdx.x >> 5;
  int m0 = blockIdx.x * 128 + (w >> 2) * 64;
  int n0 = blockIdx.y * 128 + (w & 3) * 32;
  int z = blockIdx.z;
  const unsigned short* W = (z == 0) ? wq : ((z == 1) ? wk : wv);

  v8f acc[4][2];
  gemm_core(xb, W, D_, m0, n0, lane, acc);

  int g = lane >> 4, nl = lane & 15;
#pragma unroll
  for (int i = 0; i < 4; ++i)
#pragma unroll
    for (int j = 0; j < 2; ++j)
#pragma unroll
      for (int r = 0; r < 8; ++r) {
        int m = m0 + i * 16 + r + 8 * g;      // global row (b*S + s)
        int n = n0 + j * 16 + nl;             // global col (h*Hd + hd)
        int b = m >> 11, s = m & (S_ - 1);
        int h = n >> 6, hd = n & (HD_ - 1);
        unsigned short val = f32_to_bf16(acc[i][j][r]);
        if (z == 2) {
          vt[((size_t)((b * H_ + h) * HD_ + hd)) * S_ + s] = val;
        } else {
          unsigned short* o = (z == 0) ? qb : kb;
          o[((size_t)((b * H_ + h) * S_ + s)) * HD_ + hd] = val;
        }
      }
}

// ---------------------------------------------------------------------------
// Flash attention (causal + key padding mask).
// One wave per 32 query rows: 2 M-fragment rows, K/V tiles loaded once and
// used by both -> 16 wmma per 16 b128 loads per key tile.
// grid (S/32, H, B), block 32.
// ---------------------------------------------------------------------------
__global__ __launch_bounds__(32) void
k_flash(const unsigned short* __restrict__ qb,
        const unsigned short* __restrict__ kbuf,
        const unsigned short* __restrict__ vt,
        const int* __restrict__ mask,
        unsigned short* __restrict__ ob) {
  __shared__ __align__(16) unsigned short ldsP[32 * 32];   // 2 KB

  int lane = threadIdx.x & 31;
  int g = lane >> 4, nl = lane & 15;
  int q0 = blockIdx.x * 32;
  int h = blockIdx.y;
  int b = blockIdx.z;

  const unsigned short* Qbh = qb   + ((size_t)(b * H_ + h)) * S_ * HD_;
  const unsigned short* Kbh = kbuf + ((size_t)(b * H_ + h)) * S_ * HD_;
  const unsigned short* Vbh = vt   + ((size_t)(b * H_ + h)) * HD_ * S_;
  const int* maskb = mask + b * S_;

  // Q tile 32x64 = 2 m-tiles x 2 k-halves of A-fragments
  v16bf qf[2][2];
#pragma unroll
  for (int i = 0; i < 2; ++i) {
    qf[i][0] = load_frag_nt(Qbh, HD_, q0 + i * 16, 0,  lane);
    qf[i][1] = load_frag_nt(Qbh, HD_, q0 + i * 16, 32, lane);
  }

  v8f o[2][4];
  float mrow[2][8], lrow[2][8];
#pragma unroll
  for (int i = 0; i < 2; ++i) {
#pragma unroll
    for (int f = 0; f < 4; ++f) o[i][f] = vzero8();
#pragma unroll
    for (int r = 0; r < 8; ++r) { mrow[i][r] = NEG_INF; lrow[i][r] = 0.f; }
  }

  int ntiles = blockIdx.x + 1;   // causal: key tiles of 32, keys <= q0+31

  for (int t = 0; t < ntiles; ++t) {
    int kb0 = t * 32;

    if (t + 1 < ntiles) {
      // near-scope prefetch of next K/V key tile
      __builtin_prefetch(Kbh + ((size_t)(kb0 + 32 + nl)) * HD_, 0, 3);
      __builtin_prefetch(Vbh + ((size_t)(nl)) * S_ + kb0 + 32, 0, 3);
    }

    // ---- S = Q K^T over 32 keys: 2x2 C tiles of 16x16 ---------------------
    v8f sc[2][2];
    {
      v16bf k00 = load_frag_nt(Kbh, HD_, kb0,      0,  lane);
      v16bf k01 = load_frag_nt(Kbh, HD_, kb0,      32, lane);
      v16bf k10 = load_frag_nt(Kbh, HD_, kb0 + 16, 0,  lane);
      v16bf k11 = load_frag_nt(Kbh, HD_, kb0 + 16, 32, lane);
#pragma unroll
      for (int i = 0; i < 2; ++i) {
        sc[i][0] = vzero8();
        sc[i][1] = vzero8();
        sc[i][0] = WMMA_BF16(qf[i][0], k00, sc[i][0]);
        sc[i][0] = WMMA_BF16(qf[i][1], k01, sc[i][0]);
        sc[i][1] = WMMA_BF16(qf[i][0], k10, sc[i][1]);
        sc[i][1] = WMMA_BF16(qf[i][1], k11, sc[i][1]);
      }
    }

    // ---- mask + online softmax -------------------------------------------
    int n0g = kb0 + nl;
    int n1g = kb0 + 16 + nl;
    int mv0 = maskb[n0g];
    int mv1 = maskb[n1g];

#pragma unroll
    for (int i = 0; i < 2; ++i) {
      float p0[8], p1[8];
#pragma unroll
      for (int r = 0; r < 8; ++r) {
        int mg = q0 + i * 16 + r + 8 * g;
        float s0 = sc[i][0][r] * 0.125f;
        float s1 = sc[i][1][r] * 0.125f;
        s0 = (n0g <= mg && mv0 != 0) ? s0 : NEG_INF;
        s1 = (n1g <= mg && mv1 != 0) ? s1 : NEG_INF;

        float vmax = fmaxf(s0, s1);
#pragma unroll
        for (int off = 8; off >= 1; off >>= 1)
          vmax = fmaxf(vmax, __shfl_xor(vmax, off, 16));

        float mnew = fmaxf(mrow[i][r], vmax);
        float alpha = __expf(mrow[i][r] - mnew);
        float e0 = __expf(s0 - mnew);
        float e1 = __expf(s1 - mnew);
        float rs = e0 + e1;
#pragma unroll
        for (int off = 8; off >= 1; off >>= 1)
          rs += __shfl_xor(rs, off, 16);

        lrow[i][r] = lrow[i][r] * alpha + rs;
        mrow[i][r] = mnew;
#pragma unroll
        for (int f = 0; f < 4; ++f) o[i][f][r] *= alpha;
        p0[r] = e0;
        p1[r] = e1;
      }
      // write this m-tile's P rows into LDS (C-layout f32 -> row-major bf16)
#pragma unroll
      for (int r = 0; r < 8; ++r) {
        ldsP[(i * 16 + r + 8 * g) * 32 + nl]      = f32_to_bf16(p0[r]);
        ldsP[(i * 16 + r + 8 * g) * 32 + 16 + nl] = f32_to_bf16(p1[r]);
      }
    }

    __syncthreads();
    v16bf pf0 = load_frag_nt(&ldsP[0], 32, 0,  0, lane);
    v16bf pf1 = load_frag_nt(&ldsP[0], 32, 16, 0, lane);
    __syncthreads();

    // ---- O += P * V  (V stored transposed [Hd, S] -> NT B fragments) ------
#pragma unroll
    for (int f = 0; f < 4; ++f) {
      v16bf vf = load_frag_nt(Vbh, S_, f * 16, kb0, lane);
      o[0][f] = WMMA_BF16(pf0, vf, o[0][f]);
      o[1][f] = WMMA_BF16(pf1, vf, o[1][f]);
    }
  }

  // ---- epilogue: O /= l, write bf16 [B,S,D] -------------------------------
#pragma unroll
  for (int i = 0; i < 2; ++i)
#pragma unroll
    for (int f = 0; f < 4; ++f)
#pragma unroll
      for (int r = 0; r < 8; ++r) {
        float val = o[i][f][r] / lrow[i][r];
        int mg = q0 + i * 16 + r + 8 * g;
        int col = h * HD_ + f * 16 + nl;
        ob[((size_t)(b * S_ + mg)) * D_ + col] = f32_to_bf16(val);
      }
}

// ---------------------------------------------------------------------------
// Output projection: out_f32[M,D] = Ob[M,D] * Wo[D,D]^T
// grid (M/128, D/128), block 256.
// ---------------------------------------------------------------------------
__global__ __launch_bounds__(256) void
k_out_gemm(const unsigned short* __restrict__ ob,
           const unsigned short* __restrict__ wo,
           float* __restrict__ out) {
  int lane = threadIdx.x & 31;
  int w = threadIdx.x >> 5;
  int m0 = blockIdx.x * 128 + (w >> 2) * 64;
  int n0 = blockIdx.y * 128 + (w & 3) * 32;

  v8f acc[4][2];
  gemm_core(ob, wo, D_, m0, n0, lane, acc);

  int g = lane >> 4, nl = lane & 15;
#pragma unroll
  for (int i = 0; i < 4; ++i)
#pragma unroll
    for (int j = 0; j < 2; ++j)
#pragma unroll
      for (int r = 0; r < 8; ++r) {
        int m = m0 + i * 16 + r + 8 * g;
        int n = n0 + j * 16 + nl;
        out[(size_t)m * D_ + n] = acc[i][j][r];
      }
}

// ---------------------------------------------------------------------------
// Launch
// ---------------------------------------------------------------------------
extern "C" void kernel_launch(void* const* d_in, const int* in_sizes, int n_in,
                              void* d_out, int out_size, void* d_ws, size_t ws_size,
                              hipStream_t stream) {
  (void)in_sizes; (void)n_in; (void)out_size; (void)ws_size;

  const float* x   = (const float*)d_in[0];
  const int* mask  = (const int*)d_in[1];
  const float* Wq  = (const float*)d_in[2];
  const float* Wk  = (const float*)d_in[3];
  const float* Wv  = (const float*)d_in[4];
  const float* Wo  = (const float*)d_in[5];
  float* out = (float*)d_out;

  char* ws = (char*)d_ws;
  const size_t xbBytes = (size_t)M_ * D_ * 2;            // 8 MB
  const size_t wBytes  = (size_t)D_ * D_ * 2;            // 2 MB
  const size_t qBytes  = (size_t)B_ * H_ * S_ * HD_ * 2; // 8 MB

  unsigned short* xb  = (unsigned short*)(ws);
  unsigned short* wqb = (unsigned short*)(ws + xbBytes);
  unsigned short* wkb = (unsigned short*)(ws + xbBytes + wBytes);
  unsigned short* wvb = (unsigned short*)(ws + xbBytes + 2 * wBytes);
  unsigned short* wob = (unsigned short*)(ws + xbBytes + 3 * wBytes);
  unsigned short* qb  = (unsigned short*)(ws + xbBytes + 4 * wBytes);
  unsigned short* kbb = (unsigned short*)(ws + xbBytes + 4 * wBytes + qBytes);
  unsigned short* vtb = (unsigned short*)(ws + xbBytes + 4 * wBytes + 2 * qBytes);
  unsigned short* obb = (unsigned short*)(ws + xbBytes + 4 * wBytes + 3 * qBytes);

  const int nX = M_ * D_;      // 4,194,304
  const int nW = D_ * D_;      // 1,048,576

  k_f32_to_bf16<<<(nX + 255) / 256, 256, 0, stream>>>(x,  xb,  nX);
  k_f32_to_bf16<<<(nW + 255) / 256, 256, 0, stream>>>(Wq, wqb, nW);
  k_f32_to_bf16<<<(nW + 255) / 256, 256, 0, stream>>>(Wk, wkb, nW);
  k_f32_to_bf16<<<(nW + 255) / 256, 256, 0, stream>>>(Wv, wvb, nW);
  k_f32_to_bf16<<<(nW + 255) / 256, 256, 0, stream>>>(Wo, wob, nW);

  k_qkv_gemm<<<dim3(M_ / 128, D_ / 128, 3), 256, 0, stream>>>(
      xb, wqb, wkb, wvb, qb, kbb, vtb);

  k_flash<<<dim3(S_ / 32, H_, B_), 32, 0, stream>>>(qb, kbb, vtb, mask, obb);

  k_out_gemm<<<dim3(M_ / 128, D_ / 128, 1), 256, 0, stream>>>(obb, wob, out);
}
